// PoincareConcatLinear_27685359190129
// MI455X (gfx1250) — matrix operand order repaired and checked by reading, not compiled, not run on hardware
//
#include <hip/hip_runtime.h>
#include <math.h>

#define S_STACKS 8
#define D_DIM    128
#define K_DIM    1024   // S*D
#define OUT_DIM  1024

// GEMM tiling
#define BK          32
#define A_STRIDE    80                  // bytes per staged row: 64B data + 16B pad (bank-conflict free)
#define A_BYTES     (128 * A_STRIDE)    // 10240
#define B_BYTES     (256 * A_STRIDE)    // 20480
#define STAGE_BYTES (A_BYTES + B_BYTES) // 30720 ; x2 stages = 60KB LDS

typedef __attribute__((ext_vector_type(16))) _Float16 v16h;
typedef __attribute__((ext_vector_type(8)))  float    v8f;
typedef __attribute__((ext_vector_type(4)))  _Float16 v4h;

union FragU { v16h h; float4 f4[2]; };

__device__ __forceinline__ float wave_allsum(float v) {
#pragma unroll
  for (int off = 16; off > 0; off >>= 1) v += __shfl_xor(v, off, 32);
  return v;
}

// CDNA5 async copy: global -> LDS, 16 bytes, tracked by ASYNCcnt.
// ldsOff: byte offset within the block's LDS allocation (single __shared__
// buffer => allocation base 0). The LDS buffer's address is escaped once per
// kernel (see k_gemm) so these asm statements are treated as potential writers.
__device__ __forceinline__ void async_b128(unsigned ldsOff, const void* g) {
  asm volatile("global_load_async_to_lds_b128 %0, %1, off"
               :: "v"(ldsOff),
                  "v"((unsigned long long)(uintptr_t)g));
}

// Issue one BK=32 stage of async copies: A 128x64B + B 256x64B (6 x b128/thread).
__device__ __forceinline__ void issue_stage(
    const _Float16* yh, const _Float16* wvT,
    int tid, int rowBlock, int colBlock, int k0, unsigned buf)
{
#pragma unroll
  for (int j = 0; j < 2; ++j) {          // A tile
    int idx = tid + 256 * j;
    int r = idx >> 2, q = idx & 3;
    async_b128(buf + (unsigned)(r * A_STRIDE + q * 16),
               yh + (size_t)(rowBlock + r) * K_DIM + k0 + q * 8);
  }
#pragma unroll
  for (int j = 0; j < 4; ++j) {          // B tile
    int idx = tid + 256 * j;
    int r = idx >> 2, q = idx & 3;
    async_b128(buf + A_BYTES + (unsigned)(r * A_STRIDE + q * 16),
               wvT + (size_t)(colBlock + r) * K_DIM + k0 + q * 8);
  }
}

// ---------------- Kernel 1: weight prep ----------------
__global__ __launch_bounds__(256) void k_weights(
    const float* __restrict__ wg, const float* __restrict__ wv,
    const float* __restrict__ bias, _Float16* __restrict__ wvT,
    float* __restrict__ chv, float* __restrict__ shv, float* __restrict__ g2v)
{
  __shared__ float red[256];
  const int o = blockIdx.x;      // output column
  const int t = threadIdx.x;
  float s = 0.0f;
  for (int k = t; k < K_DIM; k += 256) {
    float w = wv[(size_t)k * OUT_DIM + o];
    s += w * w;
  }
  red[t] = s;
  __syncthreads();
#pragma unroll
  for (int off = 128; off > 0; off >>= 1) {
    if (t < off) red[t] += red[t + off];
    __syncthreads();
  }
  const float inv = 1.0f / fmaxf(sqrtf(red[0]), 1e-15f);
  for (int k = t; k < K_DIM; k += 256)
    wvT[(size_t)o * K_DIM + k] = (_Float16)(wv[(size_t)k * OUT_DIM + o] * inv);
  if (t == 0) {
    float b2 = 2.0f * bias[o];
    chv[o] = coshf(b2);
    shv[o] = sinhf(b2);
    g2v[o] = 2.0f * wg[o];
  }
}

// ---------------- Kernel 2: token prep (logmap0 -> concat -> expmap0) -------
__global__ __launch_bounds__(256) void k_token(
    const float* __restrict__ x, _Float16* __restrict__ yh,
    float* __restrict__ cx2, float beta_ratio, int ntok)
{
  const int tok  = (blockIdx.x * 256 + threadIdx.x) >> 5;
  const int lane = threadIdx.x & 31;
  if (tok >= ntok) return;
  const float* xr = x + (size_t)tok * K_DIM;

  float4 v[S_STACKS];
  float  scl[S_STACKS];
  float  un2 = 0.0f;
#pragma unroll
  for (int s = 0; s < S_STACKS; ++s) {
    v[s] = *(const float4*)(xr + s * D_DIM + lane * 4);
    float ss = v[s].x * v[s].x + v[s].y * v[s].y + v[s].z * v[s].z + v[s].w * v[s].w;
    ss = wave_allsum(ss);
    float xn = fmaxf(sqrtf(ss), 1e-15f);
    float z  = fminf(xn, 1.0f - 1e-7f);
    float t  = atanhf(z) * beta_ratio;   // = ||u_s|| after beta rescale (RC=1)
    un2 += t * t;
    scl[s] = t / xn;
  }
  const float un = fmaxf(sqrtf(un2), 1e-15f);
  const float th = tanhf(un);
  const float g  = th / un;              // expmap0 radial factor
#pragma unroll
  for (int s = 0; s < S_STACKS; ++s) {
    float fs = g * scl[s];
    v4h o;
    o.x = (_Float16)(fs * v[s].x);
    o.y = (_Float16)(fs * v[s].y);
    o.z = (_Float16)(fs * v[s].z);
    o.w = (_Float16)(fs * v[s].w);
    *(v4h*)(yh + (size_t)tok * K_DIM + s * D_DIM + lane * 4) = o;
  }
  if (lane == 0) cx2[tok] = th * th;     // c*||y||^2
}

// ---------------- Kernel 3: async-LDS double-buffered WMMA GEMM ------------
// Block: 8 waves (2 M x 4 N) -> 128x256 tile; wave: 64x64 via 16 f32 accs.
#define WMMA_STEP(C, AF, BF) \
  C = __builtin_amdgcn_wmma_f32_16x16x32_f16(false, (AF).h, false, (BF).h, \
                                             (short)0, C, false, false)
#define LOADA(F, MI) do { \
    const char* p_ = aB + (waveM * 64 + (MI) * 16 + m) * A_STRIDE + half * 16; \
    (F).f4[0] = *(const float4*)(p_); \
    (F).f4[1] = *(const float4*)(p_ + 32); } while (0)
#define LOADB(F, NI) do { \
    const char* p_ = bB + (waveN * 64 + (NI) * 16 + m) * A_STRIDE + half * 16; \
    (F).f4[0] = *(const float4*)(p_); \
    (F).f4[1] = *(const float4*)(p_ + 32); } while (0)

__global__ __launch_bounds__(256, 1) void k_gemm(
    const _Float16* __restrict__ yh, const _Float16* __restrict__ wvT,
    const float* __restrict__ cx2, const float* __restrict__ chv,
    const float* __restrict__ shv, const float* __restrict__ g2v,
    float* __restrict__ out, int ntok)
{
  __shared__ uint4 smem4[2 * STAGE_BYTES / 16];
  char* smem = (char*)smem4;
  if (ntok <= 0) return;                 // uniform

  // Escape the LDS allocation's address ONCE so the async asm statements are
  // treated as potential writers of it (prevents folding the ds reads).
  asm volatile("" :: "v"((unsigned long long)(uintptr_t)smem) : "memory");

  const int tid   = threadIdx.x;
  const int lane  = tid & 31;
  const int wave  = tid >> 5;
  const int waveM = wave >> 2, waveN = wave & 3;
  const int rowBlock = blockIdx.y * 128;
  const int colBlock = blockIdx.x * 256;
  const int m = lane & 15, half = lane >> 4;

  // 16 named accumulators (plain SSA values -> stay in VGPRs).
  v8f c00 = {}, c01 = {}, c02 = {}, c03 = {};
  v8f c10 = {}, c11 = {}, c12 = {}, c13 = {};
  v8f c20 = {}, c21 = {}, c22 = {}, c23 = {};
  v8f c30 = {}, c31 = {}, c32 = {}, c33 = {};

  issue_stage(yh, wvT, tid, rowBlock, colBlock, 0, 0u);

  const int NSTAGE = K_DIM / BK;         // 32
  for (int s = 0; s < NSTAGE; ++s) {
    if (s + 1 < NSTAGE) {
      issue_stage(yh, wvT, tid, rowBlock, colBlock,
                  (s + 1) * BK, (unsigned)(((s + 1) & 1) * STAGE_BYTES));
      // 12 outstanding; async loads complete in order -> <=6 left means stage s landed.
      asm volatile("s_wait_asynccnt 0x6" ::: "memory");
    } else {
      asm volatile("s_wait_asynccnt 0x0" ::: "memory");
    }
    __syncthreads();                     // stage s visible to all waves

    const char* aB = smem + (unsigned)((s & 1) * STAGE_BYTES);
    const char* bB = aB + A_BYTES;
    // A 16-bit 16x32 layout: lane(M=m, half); f4[0]: K=half*8..+7 ; f4[1]: K=16+half*8..+7
    // B 32x16 layout: lane(N=m, half); f4[0]: K=half*8..+7 ; f4[1]: K=16+half*8..+7
    FragU a0, a1, a2, a3, b0, b1, b2, b3;
    LOADA(a0, 0); LOADA(a1, 1); LOADA(a2, 2); LOADA(a3, 3);
    LOADB(b0, 0); LOADB(b1, 1); LOADB(b2, 2); LOADB(b3, 3);

    WMMA_STEP(c00, a0, b0); WMMA_STEP(c01, a0, b1);
    WMMA_STEP(c02, a0, b2); WMMA_STEP(c03, a0, b3);
    WMMA_STEP(c10, a1, b0); WMMA_STEP(c11, a1, b1);
    WMMA_STEP(c12, a1, b2); WMMA_STEP(c13, a1, b3);
    WMMA_STEP(c20, a2, b0); WMMA_STEP(c21, a2, b1);
    WMMA_STEP(c22, a2, b2); WMMA_STEP(c23, a2, b3);
    WMMA_STEP(c30, a3, b0); WMMA_STEP(c31, a3, b1);
    WMMA_STEP(c32, a3, b2); WMMA_STEP(c33, a3, b3);

    __syncthreads();                     // protect buffer reuse at stage s+2
  }

  // Gather for the epilogue (constant indices only; no asm in between).
  v8f acc[4][4] = {{c00, c01, c02, c03},
                   {c10, c11, c12, c13},
                   {c20, c21, c22, c23},
                   {c30, c31, c32, c33}};

  // Epilogue. C/D layout: VGPR r -> M = r + 8*half ; N = lane&15.
  float chL[4], shL[4], gL[4];
  int   colL[4];
#pragma unroll
  for (int ni = 0; ni < 4; ++ni) {
    int c = colBlock + waveN * 64 + ni * 16 + m;
    colL[ni] = c; chL[ni] = chv[c]; shL[ni] = shv[c]; gL[ni] = g2v[c];
  }
#pragma unroll
  for (int mi = 0; mi < 4; ++mi) {
#pragma unroll
    for (int r = 0; r < 8; ++r) {
      int   row = rowBlock + waveM * 64 + mi * 16 + r + 8 * half;
      float cx  = cx2[row];
      float den = fmaxf(1.0f - cx, 1e-15f);
      float opc = 1.0f + cx;
#pragma unroll
      for (int ni = 0; ni < 4; ++ni) {
        float dot = acc[mi][ni][r];
        float num = 2.0f * dot * chL[ni] - opc * shL[ni];
        float hh  = gL[ni] * asinhf(num / den);
        if (!(fabsf(hh) < 3.0e38f)) hh = 0.0f;       // NaN/Inf -> 0
        hh = sinhf(fminf(fmaxf(hh, -20.0f), 20.0f));
        if (!(fabsf(hh) < 3.0e38f)) hh = 0.0f;
        out[(size_t)row * OUT_DIM + colL[ni]] = hh;
      }
    }
  }
}

// ---------------- Kernel 4: deterministic per-row sum of squares -----------
__global__ __launch_bounds__(256) void k_rownorm(
    const float* __restrict__ out, float* __restrict__ n2, int ntok)
{
  const int row  = (blockIdx.x * 256 + threadIdx.x) >> 5;
  const int lane = threadIdx.x & 31;
  if (row >= ntok) return;
  const float* hr = out + (size_t)row * OUT_DIM;
  float s = 0.0f;
#pragma unroll
  for (int c = 0; c < 8; ++c) {
    float4 v = *(const float4*)(hr + c * 128 + lane * 4);
    s += v.x * v.x + v.y * v.y + v.z * v.z + v.w * v.w;
  }
  s = wave_allsum(s);
  if (lane == 0) n2[row] = fminf(s, 1e15f);
}

// ---------------- Kernel 5: projection back into the ball ------------------
__global__ __launch_bounds__(256) void k_final(
    float* __restrict__ out, const float* __restrict__ n2, int total)
{
  const int i = blockIdx.x * 256 + threadIdx.x;
  if (i >= total) return;
  const int row = i >> 10;               // OUT_DIM == 1024
  float v = out[i];
  float d = fmaxf(1.0f + sqrtf(1.0f + n2[row]), 1e-15f);
  float r = v / d;
  if (!(fabsf(r) < 3.0e38f)) r = 0.0f;
  out[i] = r;
}

extern "C" void kernel_launch(void* const* d_in, const int* in_sizes, int n_in,
                              void* d_out, int out_size, void* d_ws, size_t ws_size,
                              hipStream_t stream) {
  const float* x    = (const float*)d_in[0];  // [N, S, D]
  const float* wg   = (const float*)d_in[1];  // [OUT]
  const float* wv   = (const float*)d_in[2];  // [K, OUT]
  const float* bias = (const float*)d_in[3];  // [OUT]
  float* out = (float*)d_out;
  const int ntok = in_sizes[0] / K_DIM;

  // Workspace carve-up (256B aligned).
  char*  ws  = (char*)d_ws;
  size_t off = 0;
  auto carve = [&](size_t bytes) {
    void* p = ws + off;
    off = (off + bytes + 255) & ~(size_t)255;
    return p;
  };
  _Float16* wvT = (_Float16*)carve((size_t)OUT_DIM * K_DIM * sizeof(_Float16));
  float*    chv = (float*)carve(OUT_DIM * sizeof(float));
  float*    shv = (float*)carve(OUT_DIM * sizeof(float));
  float*    g2v = (float*)carve(OUT_DIM * sizeof(float));
  float*    cx2 = (float*)carve((size_t)ntok * sizeof(float));
  float*    n2  = (float*)carve((size_t)ntok * sizeof(float));
  _Float16* yh  = (_Float16*)carve((size_t)ntok * K_DIM * sizeof(_Float16));
  (void)ws_size;

  // beta(S*D/2, 1/2) / beta(D/2, 1/2), computed exactly on host.
  double lr = (lgamma(512.0) - lgamma(512.5)) - (lgamma(64.0) - lgamma(64.5));
  float beta_ratio = (float)exp(lr);

  hipLaunchKernelGGL(k_weights, dim3(OUT_DIM), dim3(256), 0, stream,
                     wg, wv, bias, wvT, chv, shv, g2v);
  hipLaunchKernelGGL(k_token, dim3((ntok + 7) / 8), dim3(256), 0, stream,
                     x, yh, cx2, beta_ratio, ntok);
  hipLaunchKernelGGL(k_gemm, dim3(OUT_DIM / 256, (ntok + 127) / 128), dim3(256), 0, stream,
                     yh, wvT, cx2, chv, shv, g2v, out, ntok);
  hipLaunchKernelGGL(k_rownorm, dim3((ntok + 7) / 8), dim3(256), 0, stream,
                     out, n2, ntok);
  hipLaunchKernelGGL(k_final, dim3((ntok * OUT_DIM + 255) / 256), dim3(256), 0, stream,
                     out, n2, ntok * OUT_DIM);
}
